// HMM_VAE_67001489817712
// MI455X (gfx1250) — compile-verified, matrix operand order checked by reading, not run on hardware
//
#include <hip/hip_runtime.h>
#include <hip/hip_bf16.h>
#include <math.h>

// ---------------------------------------------------------------------------
// HMM-VAE forward for MI455X (gfx1250, wave32, WMMA 16x16x32 bf16)
// B=32, T=128, K=16, L=16, N=4096, D=784, H=400
// All GEMM operands pre-packed into bf16 fragment-layout tiles (32B/lane ->
// 2x global_load_b128 per fragment, zero hot-loop converts/branches).
// k_dec_bce stages its activation tile via async-to-LDS (when available)
// and prefetches the dec_w fragment stream.
// ---------------------------------------------------------------------------

#define NTOK   4096      // B*T
#define DIN    784
#define HID    400
#define KMIX   16
#define LLAT   16
#define BB     32
#define TT     128

typedef __bf16 v16bf __attribute__((ext_vector_type(16)));
typedef float  v8f   __attribute__((ext_vector_type(8)));

#if __has_builtin(__builtin_amdgcn_global_load_async_to_lds_b128)
#define HAVE_ASYNC_B128 1
typedef int v4i __attribute__((vector_size(16)));
typedef __attribute__((address_space(1))) v4i* gv4i_p;   // global src
typedef __attribute__((address_space(3))) v4i* lv4i_p;   // LDS dst
#endif

__device__ __forceinline__ void wait_async0() {
#if __has_builtin(__builtin_amdgcn_s_wait_asynccnt)
    __builtin_amdgcn_s_wait_asynccnt(0);
#else
    asm volatile("s_wait_asynccnt 0x0" ::: "memory");
#endif
}

// float -> bf16, round-to-nearest-even, via bit ops
__device__ __forceinline__ __bf16 f2bf(float f) {
    union { float f; unsigned u; } x; x.f = f;
    unsigned r = x.u + 0x7FFFu + ((x.u >> 16) & 1u);
    unsigned short h = (unsigned short)(r >> 16);
    return __builtin_bit_cast(__bf16, h);
}
__device__ __forceinline__ __bf16 bf_zero() {
    return __builtin_bit_cast(__bf16, (unsigned short)0);
}

// Load one packed fragment: 16 bf16 (32 bytes) per lane, lane-major tiles.
__device__ __forceinline__ v16bf load_packed(const __bf16* __restrict__ p, int tile) {
    return *(const v16bf*)(p + ((size_t)tile * 32 + (threadIdx.x & 31)) * 16);
}

__device__ __forceinline__ float softplus_f(float x) {
    float ax = fabsf(x);
    return fmaxf(x, 0.0f) + log1pf(__expf(-ax));
}

// ---------------------------------------------------------------------------
// Pack a row-major f32 matrix (rtiles*16 rows, leading dim ld, logical inner
// dim `inner`) into bf16 WMMA fragment layout tiles, tile id = rt*kblocks+kb.
// Per ISA 16-bit layout: lane (m=lane&15, hf=lane>>4) holds
// K = kb*32 + 8*hf + {0..7} and K = kb*32 + 16 + 8*hf + {0..7}.
// ---------------------------------------------------------------------------
__global__ void k_pack(const float* __restrict__ src, __bf16* __restrict__ dst,
                       int rtiles, int kblocks, int ld, int inner) {
    int wid = threadIdx.x >> 5;
    int t = blockIdx.x * 8 + wid;
    if (t >= rtiles * kblocks) return;
    int rt = t / kblocks, kb = t % kblocks;
    int lane = threadIdx.x & 31;
    int m = lane & 15, hf = lane >> 4;
    const float* p = src + (size_t)(rt * 16 + m) * ld;
    v16bf a;
    if (kb * 32 + 32 <= inner) {           // full block: 4x float4 loads
        const float* q = p + kb * 32 + 8 * hf;
        float4 u0 = *(const float4*)(q);
        float4 u1 = *(const float4*)(q + 4);
        float4 u2 = *(const float4*)(q + 16);
        float4 u3 = *(const float4*)(q + 20);
        a[0] = f2bf(u0.x); a[1] = f2bf(u0.y); a[2]  = f2bf(u0.z); a[3]  = f2bf(u0.w);
        a[4] = f2bf(u1.x); a[5] = f2bf(u1.y); a[6]  = f2bf(u1.z); a[7]  = f2bf(u1.w);
        a[8] = f2bf(u2.x); a[9] = f2bf(u2.y); a[10] = f2bf(u2.z); a[11] = f2bf(u2.w);
        a[12] = f2bf(u3.x); a[13] = f2bf(u3.y); a[14] = f2bf(u3.z); a[15] = f2bf(u3.w);
    } else {                               // tail block: branchless clamp+select
#pragma unroll
        for (int e = 0; e < 16; ++e) {
            int kk = kb * 32 + (e < 8 ? 8 * hf + e : 8 * hf + e + 8);
            int kc = kk < inner ? kk : 0;
            float f = p[kc];
            a[e] = f2bf(kk < inner ? f : 0.0f);
        }
    }
    *(v16bf*)(dst + ((size_t)t * 32 + lane) * 16) = a;
}

// ---------------------------------------------------------------------------
// lp = log(softmax(log_pi)+1e-9), lA = log(softmax(log_A,axis=1)+1e-9)
// ---------------------------------------------------------------------------
__global__ void k_prep(const float* __restrict__ log_pi,
                       const float* __restrict__ log_A,
                       float* __restrict__ lp, float* __restrict__ lA) {
    int i = threadIdx.x;
    if (i < KMIX) {
        float m = -1e30f;
        for (int j = 0; j < KMIX; ++j) m = fmaxf(m, log_A[i * KMIX + j]);
        float s = 0.0f;
        for (int j = 0; j < KMIX; ++j) s += __expf(log_A[i * KMIX + j] - m);
        for (int j = 0; j < KMIX; ++j)
            lA[i * KMIX + j] = __logf(__expf(log_A[i * KMIX + j] - m) / s + 1e-9f);
    } else if (i == KMIX) {
        float m = -1e30f;
        for (int j = 0; j < KMIX; ++j) m = fmaxf(m, log_pi[j]);
        float s = 0.0f;
        for (int j = 0; j < KMIX; ++j) s += __expf(log_pi[j] - m);
        for (int j = 0; j < KMIX; ++j)
            lp[j] = __logf(__expf(log_pi[j] - m) / s + 1e-9f);
    }
}

// ---------------------------------------------------------------------------
// h = relu(y @ W1^T + b1)   [4096 x 400], inner 784 (25 k-blocks)
// ---------------------------------------------------------------------------
__global__ void k_gemm1(const __bf16* __restrict__ Py, const __bf16* __restrict__ Pw1,
                        const float* __restrict__ b1, float* __restrict__ h) {
    int wid = threadIdx.x >> 5;
    int t = blockIdx.x * 8 + wid;          // 6400 tiles
    int nt = t / 25, ct = t % 25;
    v8f acc = {};
#pragma unroll 1
    for (int kb = 0; kb < 25; kb += 5) {   // 5-deep unroll: overlap loads/WMMA
#pragma unroll
        for (int u = 0; u < 5; ++u) {
            v16bf a = load_packed(Py,  nt * 25 + kb + u);
            v16bf b = load_packed(Pw1, ct * 25 + kb + u);
            acc = __builtin_amdgcn_wmma_f32_16x16x32_bf16(false, a, false, b,
                                                          (short)0, acc, false, false);
        }
    }
    int lane = threadIdx.x & 31;
    int c = lane & 15, hf = lane >> 4;
    float bias = b1[ct * 16 + c];
#pragma unroll
    for (int v = 0; v < 8; ++v)
        h[(size_t)(nt * 16 + v + 8 * hf) * HID + ct * 16 + c] = fmaxf(acc[v] + bias, 0.0f);
}

// ---------------------------------------------------------------------------
// ml = h @ W2^T + b2   [4096 x 32], inner 400 (13 k-blocks)
// ---------------------------------------------------------------------------
__global__ void k_gemm2(const __bf16* __restrict__ Ph, const __bf16* __restrict__ Pw2,
                        const float* __restrict__ b2, float* __restrict__ ml) {
    int wid = threadIdx.x >> 5;
    int t = blockIdx.x * 8 + wid;          // 512 tiles
    int nt = t >> 1, ct = t & 1;
    v8f acc = {};
#pragma unroll
    for (int kb = 0; kb < 13; ++kb) {
        v16bf a = load_packed(Ph,  nt * 13 + kb);
        v16bf b = load_packed(Pw2, ct * 13 + kb);
        acc = __builtin_amdgcn_wmma_f32_16x16x32_bf16(false, a, false, b,
                                                      (short)0, acc, false, false);
    }
    int lane = threadIdx.x & 31;
    int c = lane & 15, hf = lane >> 4;
    float bias = b2[ct * 16 + c];
#pragma unroll
    for (int v = 0; v < 8; ++v)
        ml[(size_t)(nt * 16 + v + 8 * hf) * 32 + ct * 16 + c] = acc[v] + bias;
}

// ---------------------------------------------------------------------------
// z = mu + eps * exp(0.5*logvar)
// ---------------------------------------------------------------------------
__global__ void k_z(const float* __restrict__ ml, const float* __restrict__ eps,
                    float* __restrict__ z) {
    int i = blockIdx.x * blockDim.x + threadIdx.x;   // N*L = 65536
    int n = i >> 4, l = i & 15;
    float mu = ml[n * 32 + l];
    float lv = ml[n * 32 + 16 + l];
    z[i] = mu + eps[i] * __expf(0.5f * lv);
}

// ---------------------------------------------------------------------------
// base = z @ Wz^T + dec_fc_b   [4096 x 400], inner 16 (one k-padded WMMA)
// ---------------------------------------------------------------------------
__global__ void k_base(const __bf16* __restrict__ Pz, const __bf16* __restrict__ Pwz,
                       const float* __restrict__ dfb, float* __restrict__ base) {
    int wid = threadIdx.x >> 5;
    int t = blockIdx.x * 8 + wid;          // 6400 tiles
    int nt = t / 25, ct = t % 25;
    v16bf a = load_packed(Pz,  nt);
    v16bf b = load_packed(Pwz, ct);
    v8f acc = {};
    acc = __builtin_amdgcn_wmma_f32_16x16x32_bf16(false, a, false, b,
                                                  (short)0, acc, false, false);
    int lane = threadIdx.x & 31;
    int c = lane & 15, hf = lane >> 4;
    float bias = dfb[ct * 16 + c];
#pragma unroll
    for (int v = 0; v < 8; ++v)
        base[(size_t)(nt * 16 + v + 8 * hf) * HID + ct * 16 + c] = acc[v] + bias;
}

// ---------------------------------------------------------------------------
// 8 relu(base+wx) values -> bf16 fragment elements [e0..e0+7], float4 LDS reads
// ---------------------------------------------------------------------------
__device__ __forceinline__ void cvt8_relu(v16bf& a, int e0,
                                          const float* __restrict__ bp,
                                          const float* __restrict__ xp) {
    float4 b0 = *(const float4*)bp;
    float4 b1 = *(const float4*)(bp + 4);
    float4 x0 = *(const float4*)xp;
    float4 x1 = *(const float4*)(xp + 4);
    a[e0 + 0] = f2bf(fmaxf(b0.x + x0.x, 0.0f));
    a[e0 + 1] = f2bf(fmaxf(b0.y + x0.y, 0.0f));
    a[e0 + 2] = f2bf(fmaxf(b0.z + x0.z, 0.0f));
    a[e0 + 3] = f2bf(fmaxf(b0.w + x0.w, 0.0f));
    a[e0 + 4] = f2bf(fmaxf(b1.x + x1.x, 0.0f));
    a[e0 + 5] = f2bf(fmaxf(b1.y + x1.y, 0.0f));
    a[e0 + 6] = f2bf(fmaxf(b1.z + x1.z, 0.0f));
    a[e0 + 7] = f2bf(fmaxf(b1.w + x1.w, 0.0f));
}

// ---------------------------------------------------------------------------
// Fused decoder + sigmoid + clipped-BCE reduction.
// One block per 16-row n-tile (256 blocks), 16 waves = 16 mixture components.
// ---------------------------------------------------------------------------
__global__ __launch_bounds__(512)
void k_dec_bce(const float* __restrict__ base, const float* __restrict__ dfw,
               const __bf16* __restrict__ Pdw, const float* __restrict__ dec_b,
               const float* __restrict__ yflat, float* __restrict__ logbk) {
    __shared__ float s_base[16 * HID];     // 25.6 KB
    __shared__ float s_wx[KMIX * HID];     // 25.6 KB, transposed: s_wx[k*HID+h]

    int n0 = blockIdx.x * 16;

    // Stage the 16x400 base tile: async b128 global->LDS when available
#ifdef HAVE_ASYNC_B128
    for (int c = threadIdx.x; c < 1600; c += 512) {        // 16B chunks
        int row = c / 100, c4 = c % 100;                   // 400 floats = 100 chunks
        const float* g = base + (size_t)(n0 + row) * HID + c4 * 4;
        __builtin_amdgcn_global_load_async_to_lds_b128(
            (gv4i_p)g, (lv4i_p)(v4i*)(s_base + c * 4), 0, 0);
    }
#else
    for (int i = threadIdx.x; i < 16 * HID; i += 512) {
        int m = i / HID, hh = i % HID;
        s_base[i] = base[(size_t)(n0 + m) * HID + hh];
    }
#endif
    for (int i = threadIdx.x; i < KMIX * HID; i += 512) {
        int k = i / HID, hh = i % HID;
        s_wx[i] = dfw[hh * 32 + LLAT + k];
    }
#ifdef HAVE_ASYNC_B128
    wait_async0();
#endif
    __syncthreads();

    int kwave = threadIdx.x >> 5;          // mixture component 0..15
    int lane  = threadIdx.x & 31;
    int m = lane & 15, hf = lane >> 4;

    // Build 13 A-fragments: relu(base + Wx[:,k]) in bf16.
    // Inner 400 = 12*32 + 16; tail: elements 0..7 valid, 8..15 zero.
    v16bf afrag[13];
    {
        const float* bp = s_base + m * HID;
        const float* xp = s_wx + kwave * HID;
#pragma unroll
        for (int kb = 0; kb < 13; ++kb) {
            v16bf a;
            int k0 = kb * 32 + 8 * hf;
            cvt8_relu(a, 0, bp + k0, xp + k0);
            if (kb < 12) {
                cvt8_relu(a, 8, bp + k0 + 16, xp + k0 + 16);
            } else {
#pragma unroll
                for (int e = 8; e < 16; ++e) a[e] = bf_zero();
            }
            afrag[kb] = a;
        }
    }

    float rec[8];
#pragma unroll
    for (int v = 0; v < 8; ++v) rec[v] = 0.0f;

#pragma unroll 1
    for (int ot = 0; ot < 49; ++ot) {      // 784/16 output-column tiles
        // prefetch next o-tile's packed dec_w fragments (global_prefetch_b8)
        if (ot + 1 < 49) {
            const __bf16* nx = Pdw + ((size_t)(ot + 1) * 13 * 32 + lane) * 16;
            __builtin_prefetch(nx, 0, 1);
        }
        v8f acc = {};
#pragma unroll
        for (int kb = 0; kb < 13; ++kb) {
            v16bf b = load_packed(Pdw, ot * 13 + kb);
            acc = __builtin_amdgcn_wmma_f32_16x16x32_bf16(false, afrag[kb], false, b,
                                                          (short)0, acc, false, false);
        }
        int o = ot * 16 + (lane & 15);
        float db = dec_b[o];
#pragma unroll
        for (int v = 0; v < 8; ++v) {
            int row = n0 + v + 8 * hf;
            float s  = acc[v] + db;
            float yv = yflat[(size_t)row * DIN + o];
            rec[v] += yv * fminf(softplus_f(-s), 100.0f)
                    + (1.0f - yv) * fminf(softplus_f(s), 100.0f);
        }
    }

    // Sum over the 16 columns held by each 16-lane half
#pragma unroll
    for (int v = 0; v < 8; ++v) {
        float r = rec[v];
        r += __shfl_xor(r, 1, 32);
        r += __shfl_xor(r, 2, 32);
        r += __shfl_xor(r, 4, 32);
        r += __shfl_xor(r, 8, 32);
        if ((lane & 15) == 0)
            logbk[(size_t)(n0 + v + 8 * hf) * KMIX + kwave] = -r * 0.01f;
    }
}

// ---------------------------------------------------------------------------
// HMM forward: alpha[b,t,j].  One wave per batch, lane&15 = state j.
// ---------------------------------------------------------------------------
__global__ void k_forward(const float* __restrict__ logbk, const float* __restrict__ lp,
                          const float* __restrict__ lA, float* __restrict__ alpha) {
    int wid = threadIdx.x >> 5;
    int b = blockIdx.x * 8 + wid;
    int lane = threadIdx.x & 31;
    int j = lane & 15;
    const float* lb = logbk + (size_t)b * TT * KMIX;
    float* al = alpha + (size_t)b * TT * KMIX;

    float a = lp[j] + lb[j];
    if (lane < 16) al[j] = a;
#pragma unroll 1
    for (int t = 1; t < TT; ++t) {
        float m = -1e30f;
#pragma unroll
        for (int i2 = 0; i2 < 16; ++i2) {
            float pv = __shfl(a, i2, 32);
            m = fmaxf(m, pv + lA[i2 * KMIX + j]);
        }
        float s = 0.0f;
#pragma unroll
        for (int i2 = 0; i2 < 16; ++i2) {
            float pv = __shfl(a, i2, 32);
            s += __expf(pv + lA[i2 * KMIX + j] - m);
        }
        a = m + __logf(s) + lb[t * KMIX + j];
        if (lane < 16) al[t * KMIX + j] = a;
    }
}

// ---------------------------------------------------------------------------
// HMM backward: beta[b,t,i].
// ---------------------------------------------------------------------------
__global__ void k_backward(const float* __restrict__ logbk, const float* __restrict__ lA,
                           float* __restrict__ beta) {
    int wid = threadIdx.x >> 5;
    int b = blockIdx.x * 8 + wid;
    int lane = threadIdx.x & 31;
    int j = lane & 15;
    const float* lb = logbk + (size_t)b * TT * KMIX;
    float* be = beta + (size_t)b * TT * KMIX;

    float bt = 0.0f;
    if (lane < 16) be[(TT - 1) * KMIX + j] = 0.0f;
#pragma unroll 1
    for (int t = TT - 2; t >= 0; --t) {
        float w = lb[(t + 1) * KMIX + j] + bt;
        float m = -1e30f;
#pragma unroll
        for (int jj = 0; jj < 16; ++jj) {
            float wv = __shfl(w, jj, 32);
            m = fmaxf(m, lA[j * KMIX + jj] + wv);
        }
        float s = 0.0f;
#pragma unroll
        for (int jj = 0; jj < 16; ++jj) {
            float wv = __shfl(w, jj, 32);
            s += __expf(lA[j * KMIX + jj] + wv - m);
        }
        bt = m + __logf(s);
        if (lane < 16) be[t * KMIX + j] = bt;
    }
}

// ---------------------------------------------------------------------------
// gamma[b,t,k] and xi[b,t,i,j].  One wave per (b,t).
// ---------------------------------------------------------------------------
__global__ void k_post(const float* __restrict__ alpha, const float* __restrict__ beta,
                       const float* __restrict__ logbk, const float* __restrict__ lA,
                       float* __restrict__ gamma, float* __restrict__ xi) {
    int wid = threadIdx.x >> 5;
    int u = blockIdx.x * 8 + wid;          // 4096 (b,t) units
    int b = u >> 7, t = u & 127;
    int lane = threadIdx.x & 31;
    const float* al = alpha + ((size_t)b * TT + t) * KMIX;
    const float* be = beta  + ((size_t)b * TT + t) * KMIX;

    int j = lane & 15;
    float lg = al[j] + be[j];
    float m = lg;
    m = fmaxf(m, __shfl_xor(m, 1, 32));
    m = fmaxf(m, __shfl_xor(m, 2, 32));
    m = fmaxf(m, __shfl_xor(m, 4, 32));
    m = fmaxf(m, __shfl_xor(m, 8, 32));
    float s = __expf(lg - m);
    s += __shfl_xor(s, 1, 32);
    s += __shfl_xor(s, 2, 32);
    s += __shfl_xor(s, 4, 32);
    s += __shfl_xor(s, 8, 32);
    float lse = m + __logf(s);
    if (lane < 16) gamma[((size_t)b * TT + t) * KMIX + j] = __expf(lg - lse);

    if (t < TT - 1) {
        const float* lbn = logbk + ((size_t)b * TT + t + 1) * KMIX;
        const float* ben = be + KMIX;
        float vals[8];
        float mx = -1e30f;
#pragma unroll
        for (int q = 0; q < 8; ++q) {
            int idx = lane + 32 * q;
            int i2 = idx >> 4, jj = idx & 15;
            float v = al[i2] + lA[i2 * KMIX + jj] + lbn[jj] + ben[jj];
            vals[q] = v;
            mx = fmaxf(mx, v);
        }
        mx = fmaxf(mx, __shfl_xor(mx, 1, 32));
        mx = fmaxf(mx, __shfl_xor(mx, 2, 32));
        mx = fmaxf(mx, __shfl_xor(mx, 4, 32));
        mx = fmaxf(mx, __shfl_xor(mx, 8, 32));
        mx = fmaxf(mx, __shfl_xor(mx, 16, 32));
        float ss = 0.0f;
#pragma unroll
        for (int q = 0; q < 8; ++q) ss += __expf(vals[q] - mx);
        ss += __shfl_xor(ss, 1, 32);
        ss += __shfl_xor(ss, 2, 32);
        ss += __shfl_xor(ss, 4, 32);
        ss += __shfl_xor(ss, 8, 32);
        ss += __shfl_xor(ss, 16, 32);
        float l2 = mx + __logf(ss);
        float* xo = xi + ((size_t)b * (TT - 1) + t) * 256;
#pragma unroll
        for (int q = 0; q < 8; ++q)
            xo[lane + 32 * q] = __expf(vals[q] - l2);
    }
}

// ---------------------------------------------------------------------------
extern "C" void kernel_launch(void* const* d_in, const int* in_sizes, int n_in,
                              void* d_out, int out_size, void* d_ws, size_t ws_size,
                              hipStream_t stream) {
    const float* y_batch  = (const float*)d_in[0];
    const float* enc_w1   = (const float*)d_in[1];
    const float* enc_b1   = (const float*)d_in[2];
    const float* enc_w2   = (const float*)d_in[3];
    const float* enc_b2   = (const float*)d_in[4];
    const float* dec_fc_w = (const float*)d_in[5];
    const float* dec_fc_b = (const float*)d_in[6];
    const float* dec_w    = (const float*)d_in[7];
    const float* dec_b    = (const float*)d_in[8];
    const float* log_pi   = (const float*)d_in[9];
    const float* log_A    = (const float*)d_in[10];
    const float* eps      = (const float*)d_in[11];

    // f32 workspace region
    float* ws = (float*)d_ws;
    float* w_h     = ws;                                // 4096*400
    float* w_base  = w_h     + (size_t)NTOK * HID;      // 4096*400
    float* w_ml    = w_base  + (size_t)NTOK * HID;      // 4096*32
    float* w_z     = w_ml    + (size_t)NTOK * 32;       // 4096*16
    float* w_logbk = w_z     + (size_t)NTOK * LLAT;     // 4096*16
    float* w_alpha = w_logbk + (size_t)NTOK * KMIX;     // 4096*16
    float* w_beta  = w_alpha + (size_t)NTOK * KMIX;     // 4096*16
    float* w_lA    = w_beta  + (size_t)NTOK * KMIX;     // 256
    float* w_lp    = w_lA    + KMIX * KMIX;             // 16

    // bf16 packed-fragment region (starts 32B-aligned after f32 region)
    __bf16* bw = (__bf16*)(w_lp + KMIX);
    __bf16* P_y  = bw;                                  // 256*25 tiles
    __bf16* P_w1 = P_y  + (size_t)256 * 25 * 512;       // 25*25
    __bf16* P_w2 = P_w1 + (size_t)25 * 25 * 512;        // 2*13
    __bf16* P_dw = P_w2 + (size_t)2 * 13 * 512;         // 49*13
    __bf16* P_wz = P_dw + (size_t)49 * 13 * 512;        // 25*1
    __bf16* P_h  = P_wz + (size_t)25 * 1 * 512;         // 256*13
    __bf16* P_z  = P_h  + (size_t)256 * 13 * 512;       // 256*1

    float* gamma = (float*)d_out;                       // B*T*K = 65536
    float* xi    = gamma + (size_t)BB * TT * KMIX;      // B*127*256

    k_prep<<<1, 64, 0, stream>>>(log_pi, log_A, w_lp, w_lA);

    // pack static operands
    k_pack<<<800, 256, 0, stream>>>(y_batch,  P_y,  256, 25, DIN, DIN);
    k_pack<<<79,  256, 0, stream>>>(enc_w1,   P_w1, 25,  25, DIN, DIN);
    k_pack<<<4,   256, 0, stream>>>(enc_w2,   P_w2, 2,   13, HID, HID);
    k_pack<<<80,  256, 0, stream>>>(dec_w,    P_dw, 49,  13, HID, HID);
    k_pack<<<4,   256, 0, stream>>>(dec_fc_w, P_wz, 25,  1,  32,  LLAT);

    k_gemm1<<<800, 256, 0, stream>>>(P_y, P_w1, enc_b1, w_h);
    k_pack <<<416, 256, 0, stream>>>(w_h, P_h, 256, 13, HID, HID);
    k_gemm2<<<64,  256, 0, stream>>>(P_h, P_w2, enc_b2, w_ml);
    k_z    <<<256, 256, 0, stream>>>(w_ml, eps, w_z);
    k_pack <<<32,  256, 0, stream>>>(w_z, P_z, 256, 1, LLAT, LLAT);
    k_base <<<800, 256, 0, stream>>>(P_z, P_wz, dec_fc_b, w_base);

    k_dec_bce<<<256, 512, 0, stream>>>(w_base, dec_fc_w, P_dw, dec_b, y_batch, w_logbk);

    k_forward <<<4, 256, 0, stream>>>(w_logbk, w_lp, w_lA, w_alpha);
    k_backward<<<4, 256, 0, stream>>>(w_logbk, w_lA, w_beta);
    k_post    <<<512, 256, 0, stream>>>(w_alpha, w_beta, w_logbk, w_lA, gamma, xi);
}
